// Attention_24412594111036
// MI455X (gfx1250) — matrix operand order repaired and checked by reading, not compiled, not run on hardware
//
#include <hip/hip_runtime.h>

// ---------------------------------------------------------------------------
// CDNA5 (gfx1250) attention+GRU decoder.
// Matmuls use v_wmma_f32_16x16x32_bf16 (bf16 inputs, f32 accumulate).
// ---------------------------------------------------------------------------

typedef __bf16 bf16_t;
typedef __attribute__((ext_vector_type(16))) __bf16 v16bf;
typedef __attribute__((ext_vector_type(8)))  float  v8f;

static constexpr int T_  = 64;
static constexpr int B_  = 512;
static constexpr int C_  = 512;
static constexpr int H_  = 512;
static constexpr int E_  = 128;
static constexpr int NC_ = 96;
static constexpr int S_  = 32;
static constexpr int CE_ = C_ + E_;   // 640
static constexpr int H3_ = 3 * H_;    // 1536

#define DEVINL __device__ __forceinline__

union BfFrag { v16bf v; uint4 q[2]; };

// A fragment (16x32 bf16, row-major source, leading dim = K):
// lane<16: M=lane, K = {0..7, 16..23}; lane>=16: M=lane-16, K = {8..15, 24..31}
DEVINL v16bf load_a_bf16(const bf16_t* __restrict__ A, int ldk, int row, int k0, int lane) {
  const bf16_t* p = A + (size_t)row * ldk + k0 + ((lane >> 4) << 3);
  BfFrag f;
  f.q[0] = *reinterpret_cast<const uint4*>(p);        // K offsets +0..7
  f.q[1] = *reinterpret_cast<const uint4*>(p + 16);   // K offsets +16..23
  return f.v;
}

DEVINL v16bf load_a_f32(const float* __restrict__ A, int ldk, int row, int k0, int lane) {
  const float* p = A + (size_t)row * ldk + k0 + ((lane >> 4) << 3);
  float4 f0 = *reinterpret_cast<const float4*>(p);
  float4 f1 = *reinterpret_cast<const float4*>(p + 4);
  float4 f2 = *reinterpret_cast<const float4*>(p + 16);
  float4 f3 = *reinterpret_cast<const float4*>(p + 20);
  v16bf a;
  a[0]  = (bf16_t)f0.x; a[1]  = (bf16_t)f0.y; a[2]  = (bf16_t)f0.z; a[3]  = (bf16_t)f0.w;
  a[4]  = (bf16_t)f1.x; a[5]  = (bf16_t)f1.y; a[6]  = (bf16_t)f1.z; a[7]  = (bf16_t)f1.w;
  a[8]  = (bf16_t)f2.x; a[9]  = (bf16_t)f2.y; a[10] = (bf16_t)f2.z; a[11] = (bf16_t)f2.w;
  a[12] = (bf16_t)f3.x; a[13] = (bf16_t)f3.y; a[14] = (bf16_t)f3.z; a[15] = (bf16_t)f3.w;
  return a;
}

// B fragment (32x16 bf16). Weight stored as W[N][K] row-major; B[k][n] = W[n][k].
// lane<16: N=lane, K = {0..7, 16..23}; lane>=16: N=lane-16, K = {8..15, 24..31}
DEVINL v16bf load_b(const bf16_t* __restrict__ W, int ldk, int col, int k0, int lane) {
  const bf16_t* p = W + (size_t)col * ldk + k0 + ((lane >> 4) << 3);
  BfFrag f;
  f.q[0] = *reinterpret_cast<const uint4*>(p);
  f.q[1] = *reinterpret_cast<const uint4*>(p + 16);
  return f.v;
}

// ---------------------------------------------------------------------------
// Generic C = A * W^T + bias.  A: [M,K] (f32 or bf16), W: [N,K] bf16,
// C: [M,N] (f32 or bf16).  Each wave computes a 32x32 supertile (2x2 WMMA
// tiles: 4 accumulators, A/B fragments reused across the pair).
// M,N multiples of 32; K multiple of 32.  8 waves / block.
// ---------------------------------------------------------------------------
template <bool A_F32, bool C_BF16>
__global__ __launch_bounds__(256)
void wmma_gemm(const void* __restrict__ Ap, const bf16_t* __restrict__ W,
               const float* __restrict__ bias, void* __restrict__ Cp,
               int M, int N, int K) {
  const int lane  = threadIdx.x & 31;
  const int wave  = threadIdx.x >> 5;
  const int nT2N  = N >> 5;
  const int nT2M  = M >> 5;
  const int nTile = nT2M * nT2N;
  int t = blockIdx.x * 8 + wave;
  if (t >= nTile) return;                 // wave-uniform: EXEC stays all-ones
  const int m0 = (t / nT2N) << 5;
  const int n0 = (t % nT2N) << 5;
  const int ar0 = m0 + (lane & 15), ar1 = ar0 + 16;
  const int bc0 = n0 + (lane & 15), bc1 = bc0 + 16;

  v8f acc00 = {}, acc01 = {}, acc10 = {}, acc11 = {};

  for (int k0 = 0; k0 < K; k0 += 32) {
    v16bf a0, a1;
    if constexpr (A_F32) {
      const float* A = (const float*)Ap;
      a0 = load_a_f32(A, K, ar0, k0, lane);
      a1 = load_a_f32(A, K, ar1, k0, lane);
    } else {
      const bf16_t* A = (const bf16_t*)Ap;
      a0 = load_a_bf16(A, K, ar0, k0, lane);
      a1 = load_a_bf16(A, K, ar1, k0, lane);
    }
    v16bf b0 = load_b(W, K, bc0, k0, lane);
    v16bf b1 = load_b(W, K, bc1, k0, lane);
    acc00 = __builtin_amdgcn_wmma_f32_16x16x32_bf16(false, a0, false, b0, (short)0, acc00, false, false);
    acc01 = __builtin_amdgcn_wmma_f32_16x16x32_bf16(false, a0, false, b1, (short)0, acc01, false, false);
    acc10 = __builtin_amdgcn_wmma_f32_16x16x32_bf16(false, a1, false, b0, (short)0, acc10, false, false);
    acc11 = __builtin_amdgcn_wmma_f32_16x16x32_bf16(false, a1, false, b1, (short)0, acc11, false, false);
  }

  // C/D layout: VGPR r, lanes 0-15 -> M=r, lanes 16-31 -> M=r+8; N = lane&15.
  const int cn0 = n0 + (lane & 15), cn1 = cn0 + 16;
  const int cm  = m0 + ((lane >> 4) << 3);
  const float bb0 = bias ? bias[cn0] : 0.f;
  const float bb1 = bias ? bias[cn1] : 0.f;
#pragma unroll
  for (int r = 0; r < 8; ++r) {
    const size_t row0 = (size_t)(cm + r) * N;
    const size_t row1 = (size_t)(cm + r + 16) * N;
    float v00 = acc00[r] + bb0, v01 = acc01[r] + bb1;
    float v10 = acc10[r] + bb0, v11 = acc11[r] + bb1;
    if constexpr (C_BF16) {
      bf16_t* C = (bf16_t*)Cp;
      C[row0 + cn0] = (bf16_t)v00;  C[row0 + cn1] = (bf16_t)v01;
      C[row1 + cn0] = (bf16_t)v10;  C[row1 + cn1] = (bf16_t)v11;
    } else {
      float* C = (float*)Cp;
      C[row0 + cn0] = v00;  C[row0 + cn1] = v01;
      C[row1 + cn0] = v10;  C[row1 + cn1] = v11;
    }
  }
}

// e[t,b] = sum_h tanh(fproj[t,b,h] + hp[b,h]) * W_score[h].  One wave per row.
__global__ __launch_bounds__(256)
void score_kernel(const bf16_t* __restrict__ fproj, const float* __restrict__ hp,
                  const float* __restrict__ wscore, float* __restrict__ e) {
  const int row  = blockIdx.x * 8 + (threadIdx.x >> 5);  // row = t*B + b
  const int lane = threadIdx.x & 31;
  const int b    = row & (B_ - 1);
  const bf16_t* fr  = fproj + (size_t)row * H_;
  const float*  hpr = hp + (size_t)b * H_;
  float s = 0.f;
#pragma unroll 4
  for (int h = lane; h < H_; h += 32)
    s += tanhf((float)fr[h] + hpr[h]) * wscore[h];
  for (int off = 16; off > 0; off >>= 1)
    s += __shfl_down(s, off, 32);
  if (lane == 0) e[row] = s;
}

// softmax over T (=64) for each b: one wave handles t=lane and t=lane+32.
__global__ __launch_bounds__(256)
void softmax_kernel(const float* __restrict__ e, float* __restrict__ alpha) {
  const int b    = blockIdx.x * 8 + (threadIdx.x >> 5);
  const int lane = threadIdx.x & 31;
  float v0 = e[lane * B_ + b];
  float v1 = e[(lane + 32) * B_ + b];
  float m = fmaxf(v0, v1);
  for (int off = 16; off > 0; off >>= 1) m = fmaxf(m, __shfl_xor(m, off, 32));
  float x0 = expf(v0 - m), x1 = expf(v1 - m);
  float ssum = x0 + x1;
  for (int off = 16; off > 0; off >>= 1) ssum += __shfl_xor(ssum, off, 32);
  float inv = 1.f / ssum;
  alpha[lane * B_ + b]        = x0 * inv;
  alpha[(lane + 32) * B_ + b] = x1 * inv;
}

// context[b,c] = sum_t alpha[t,b] * feature[t,b,c], written as bf16 into x[:, :C].
__global__ __launch_bounds__(256)
void context_kernel(const float* __restrict__ feature, const float* __restrict__ alpha,
                    bf16_t* __restrict__ xbf) {
  __shared__ float sa[T_];
  const int b = blockIdx.y;
  const int c = blockIdx.x * 256 + threadIdx.x;
  if (threadIdx.x < T_) sa[threadIdx.x] = alpha[threadIdx.x * B_ + b];
  __syncthreads();
  float acc = 0.f;
#pragma unroll 4
  for (int t = 0; t < T_; ++t)
    acc += sa[t] * feature[((size_t)t * B_ + b) * C_ + c];
  xbf[(size_t)b * CE_ + c] = (bf16_t)acc;
}

// x[:, C:C+E] = char_emb[target(step)]
__global__ __launch_bounds__(256)
void emb_kernel(const int* __restrict__ text, const float* __restrict__ char_emb,
                bf16_t* __restrict__ xbf, int step) {
  const int idx = blockIdx.x * 256 + threadIdx.x;
  if (idx >= B_ * E_) return;
  const int b = idx / E_, e = idx % E_;
  const int tgt = (step == 0) ? 0 : text[b * S_ + step - 1];
  xbf[(size_t)b * CE_ + C_ + e] = (bf16_t)char_emb[tgt * E_ + e];
}

// GRU cell pointwise update; writes f32 hidden, bf16 hidden, and hs[b,step,:].
__global__ __launch_bounds__(256)
void gru_kernel(const float* __restrict__ gi, const float* __restrict__ gh,
                float* __restrict__ hidden, bf16_t* __restrict__ hidden_bf,
                bf16_t* __restrict__ hs, int step) {
  const int idx = blockIdx.x * 256 + threadIdx.x;
  if (idx >= B_ * H_) return;
  const int b = idx >> 9;           // H_ == 512
  const int h = idx & (H_ - 1);
  const float* gib = gi + (size_t)b * H3_;
  const float* ghb = gh + (size_t)b * H3_;
  const float r = 1.f / (1.f + expf(-(gib[h]          + ghb[h])));
  const float z = 1.f / (1.f + expf(-(gib[H_ + h]     + ghb[H_ + h])));
  const float n = tanhf(gib[2 * H_ + h] + r * ghb[2 * H_ + h]);
  const float hn = (1.f - z) * n + z * hidden[idx];
  hidden[idx]    = hn;
  hidden_bf[idx] = (bf16_t)hn;
  hs[((size_t)b * S_ + step) * H_ + h] = (bf16_t)hn;
}

__global__ __launch_bounds__(256)
void init_hidden(float* __restrict__ hidden, bf16_t* __restrict__ hidden_bf) {
  const int idx = blockIdx.x * 256 + threadIdx.x;
  if (idx < B_ * H_) { hidden[idx] = 0.f; hidden_bf[idx] = (bf16_t)0.f; }
}

__global__ __launch_bounds__(256)
void f32_to_bf16(const float* __restrict__ src, bf16_t* __restrict__ dst, int n) {
  const int i = blockIdx.x * 256 + threadIdx.x;
  if (i < n) dst[i] = (bf16_t)src[i];
}

// ---------------------------------------------------------------------------

extern "C" void kernel_launch(void* const* d_in, const int* in_sizes, int n_in,
                              void* d_out, int out_size, void* d_ws, size_t ws_size,
                              hipStream_t stream) {
  const float* feature  = (const float*)d_in[0];
  const int*   text     = (const int*)  d_in[1];
  const float* W_h2h    = (const float*)d_in[2];
  const float* b_h2h    = (const float*)d_in[3];
  const float* W_c2h    = (const float*)d_in[4];
  const float* W_score  = (const float*)d_in[5];
  const float* W_ih     = (const float*)d_in[6];
  const float* W_hh     = (const float*)d_in[7];
  const float* b_ih     = (const float*)d_in[8];
  const float* b_hh     = (const float*)d_in[9];
  const float* char_emb = (const float*)d_in[10];
  const float* W_gen    = (const float*)d_in[11];
  const float* b_gen    = (const float*)d_in[12];
  float* out = (float*)d_out;

  // --- carve workspace (~65 MB) ---
  char* p = (char*)d_ws;
  auto carve = [&](size_t bytes) -> void* {
    void* r = (void*)p;
    p += (bytes + 255) & ~size_t(255);
    return r;
  };
  bf16_t* fproj_bf  = (bf16_t*)carve((size_t)T_ * B_ * H_ * 2);   // 32 MB
  bf16_t* Wc2h_bf   = (bf16_t*)carve((size_t)H_ * C_ * 2);
  bf16_t* Wh2h_bf   = (bf16_t*)carve((size_t)H_ * H_ * 2);
  bf16_t* Wih_bf    = (bf16_t*)carve((size_t)H3_ * CE_ * 2);
  bf16_t* Whh_bf    = (bf16_t*)carve((size_t)H3_ * H_ * 2);
  bf16_t* Wgen_bf   = (bf16_t*)carve((size_t)NC_ * H_ * 2);
  float*  hidden    = (float*) carve((size_t)B_ * H_ * 4);
  bf16_t* hidden_bf = (bf16_t*)carve((size_t)B_ * H_ * 2);
  float*  hp        = (float*) carve((size_t)B_ * H_ * 4);
  float*  e_buf     = (float*) carve((size_t)T_ * B_ * 4);
  float*  alpha     = (float*) carve((size_t)T_ * B_ * 4);
  bf16_t* x_bf      = (bf16_t*)carve((size_t)B_ * CE_ * 2);
  float*  gi        = (float*) carve((size_t)B_ * H3_ * 4);
  float*  gh        = (float*) carve((size_t)B_ * H3_ * 4);
  bf16_t* hs_bf     = (bf16_t*)carve((size_t)B_ * S_ * H_ * 2);   // 16 MB

  // --- weight conversion to bf16 (once per call) ---
  auto cvt = [&](const float* s, bf16_t* d, int n) {
    f32_to_bf16<<<(n + 255) / 256, 256, 0, stream>>>(s, d, n);
  };
  cvt(W_c2h, Wc2h_bf, H_ * C_);
  cvt(W_h2h, Wh2h_bf, H_ * H_);
  cvt(W_ih,  Wih_bf,  H3_ * CE_);
  cvt(W_hh,  Whh_bf,  H3_ * H_);
  cvt(W_gen, Wgen_bf, NC_ * H_);

  init_hidden<<<(B_ * H_ + 255) / 256, 256, 0, stream>>>(hidden, hidden_bf);

  auto gemm_blocks = [](int M, int N) { return ((M >> 5) * (N >> 5) + 7) / 8; };

  // fproj[t*B+b, h] = feature @ W_c2h^T   (f32 A -> bf16 C, no bias)
  wmma_gemm<true, true><<<gemm_blocks(T_ * B_, H_), 256, 0, stream>>>(
      feature, Wc2h_bf, nullptr, fproj_bf, T_ * B_, H_, C_);

  for (int s = 0; s < S_; ++s) {
    // hp = hidden @ W_h2h^T + b_h2h
    wmma_gemm<false, false><<<gemm_blocks(B_, H_), 256, 0, stream>>>(
        hidden_bf, Wh2h_bf, b_h2h, hp, B_, H_, H_);
    // e[t,b] = <tanh(fproj + hp), W_score>
    score_kernel<<<(T_ * B_) / 8, 256, 0, stream>>>(fproj_bf, hp, W_score, e_buf);
    // alpha = softmax_T(e)
    softmax_kernel<<<B_ / 8, 256, 0, stream>>>(e_buf, alpha);
    // x[:, :C] = context = sum_t alpha * feature
    context_kernel<<<dim3(C_ / 256, B_), 256, 0, stream>>>(feature, alpha, x_bf);
    // x[:, C:] = char_emb[target(s)]
    emb_kernel<<<(B_ * E_ + 255) / 256, 256, 0, stream>>>(text, char_emb, x_bf, s);
    // gi = x @ W_ih^T + b_ih ; gh = hidden @ W_hh^T + b_hh
    wmma_gemm<false, false><<<gemm_blocks(B_, H3_), 256, 0, stream>>>(
        x_bf, Wih_bf, b_ih, gi, B_, H3_, CE_);
    wmma_gemm<false, false><<<gemm_blocks(B_, H3_), 256, 0, stream>>>(
        hidden_bf, Whh_bf, b_hh, gh, B_, H3_, H_);
    // GRU update (writes hidden, hidden_bf, hs[b,s,:])
    gru_kernel<<<(B_ * H_ + 255) / 256, 256, 0, stream>>>(
        gi, gh, hidden, hidden_bf, hs_bf, s);
  }

  // probs = hs @ W_gen^T + b_gen   -> d_out [B*S, NC]
  wmma_gemm<false, false><<<gemm_blocks(B_ * S_, NC_), 256, 0, stream>>>(
      hs_bf, Wgen_bf, b_gen, out, B_ * S_, NC_, H_);
}
